// AdaptiveAttentionLayer_86938728005661
// MI455X (gfx1250) — compile-verified, matrix operand order checked by reading, not run on hardware
//
#include <hip/hip_runtime.h>
#include <hip/hip_bf16.h>
#include <math.h>

// Sizes from the reference
#define BB 4
#define NN 4096   // H*W
#define CC 512
#define C1 960

typedef __bf16 v16bf __attribute__((ext_vector_type(16)));
typedef float  v8f   __attribute__((ext_vector_type(8)));

__device__ __forceinline__ v8f wmma_bf16(v16bf a, v16bf b, v8f c) {
  // D = A(16x32 bf16) * B(32x16 bf16) + C(16x16 f32)
  return __builtin_amdgcn_wmma_f32_16x16x32_bf16(false, a, false, b, (short)0, c, false, false);
}

__device__ __forceinline__ v8f zero8() {
  v8f z = {0.f,0.f,0.f,0.f,0.f,0.f,0.f,0.f};
  return z;
}

// Async global->LDS copy of one 32x960 bf16 K tile (61440 B = 3840 x b128).
// Uses the CDNA5 async path (ASYNCcnt), bypassing VGPRs, enabling double buffering.
__device__ __forceinline__ void async_copy_ktile(const __bf16* gsrc, __bf16* lds_dst, int tid) {
  unsigned lbase = (unsigned)(uintptr_t)lds_dst;  // low 32 bits of generic ptr = LDS offset
  const char* g = (const char*)gsrc;
  for (int i = tid; i < 3840; i += 256) {
    unsigned loff = lbase + i * 16;
    unsigned long long ga = (unsigned long long)(uintptr_t)(g + (size_t)i * 16);
    asm volatile("global_load_async_to_lds_b128 %0, %1, off"
                 :: "v"(loff), "v"(ga) : "memory");
  }
}

__device__ __forceinline__ void wait_async0() {
  asm volatile("s_wait_asynccnt 0x0" ::: "memory");
}

// ---------------------------------------------------------------------------
// Stats: zero -> atomic accumulate (sum, sumsq per (b, channel)) -> finalize
// ---------------------------------------------------------------------------
__global__ void k_zero_f32(float* p, int n) {
  int i = blockIdx.x * blockDim.x + threadIdx.x;
  if (i < n) p[i] = 0.f;
}

// grid = B * 8 * CB, block = 256. Each block: (b, spatial-chunk, channel-block)
__global__ void k_accum_stats(const float* __restrict__ x, float* __restrict__ sum,
                              float* __restrict__ sum2, int Cdim, int CB) {
  int tid = threadIdx.x;
  int cb  = blockIdx.x % CB;
  int t   = blockIdx.x / CB;
  int sc  = t % 8;
  int b   = t / 8;
  int ch  = cb * 256 + tid;
  if (ch >= Cdim) return;
  const float* base = x + ((size_t)b * NN + sc * (NN / 8)) * Cdim + ch;
  float s = 0.f, s2 = 0.f;
  for (int i = 0; i < NN / 8; i++) {
    float v = base[(size_t)i * Cdim];
    s += v; s2 += v * v;
  }
  atomicAdd(&sum[b * Cdim + ch], s);
  atomicAdd(&sum2[b * Cdim + ch], s2);
}

__global__ void k_finalize_stats(const float* __restrict__ sum, const float* __restrict__ sum2,
                                 float* __restrict__ mean, float* __restrict__ rstd, int n) {
  int i = blockIdx.x * blockDim.x + threadIdx.x;
  if (i >= n) return;
  float m = sum[i] * (1.0f / NN);
  float v = sum2[i] * (1.0f / NN) - m * m;
  mean[i] = m;
  rstd[i] = rsqrtf(v + 1e-5f);
}

// Wt[d*E + e] = bf16(W[e*D + d])  (so WMMA B-tiles read contiguously)
__global__ void k_transpose_w(const float* __restrict__ W, __bf16* __restrict__ Wt, int E, int D) {
  int i = blockIdx.x * blockDim.x + threadIdx.x;
  if (i >= E * D) return;
  int e = i / D, d = i % D;
  Wt[(size_t)d * E + e] = (__bf16)W[i];
}

// ---------------------------------------------------------------------------
// Q/K projection: Out[b,n,:] = inorm(X[b,n,:]) @ W + bias   (bf16 out)
// grid = B*256 (16 rows per block), block = 256 (8 waves; each wave 16x16 tiles)
// ---------------------------------------------------------------------------
__global__ __launch_bounds__(256) void k_proj_norm_gemm(
    const float* __restrict__ X, const float* __restrict__ mean, const float* __restrict__ rstd,
    const __bf16* __restrict__ Wt, const float* __restrict__ bias, __bf16* __restrict__ Out) {
  __shared__ __align__(32) __bf16 Xs[16 * C1];
  int b = blockIdx.x >> 8;
  int row0 = (blockIdx.x & 255) * 16;
  int tid = threadIdx.x;
  const float* mb = mean + b * C1;
  const float* rb = rstd + b * C1;
  for (int idx = tid; idx < 16 * C1; idx += 256) {
    int r = idx / C1, ch = idx % C1;
    float v = X[((size_t)b * NN + row0 + r) * C1 + ch];
    Xs[idx] = (__bf16)((v - mb[ch]) * rb[ch]);
  }
  __syncthreads();
  int wave = tid >> 5, lane = tid & 31, ln = lane & 15, hi = lane >> 4;
  for (int ct = wave; ct < C1 / 16; ct += 8) {
    v8f acc = zero8();
    for (int ch = 0; ch < C1 / 32; ch++) {
      v16bf a  = *(const v16bf*)&Xs[ln * C1 + ch * 32 + hi * 16];
      v16bf bm = *(const v16bf*)&Wt[((size_t)(ct * 16 + ln)) * C1 + ch * 32 + hi * 16];
      acc = wmma_bf16(a, bm, acc);
    }
    float bv = bias[ct * 16 + ln];
#pragma unroll
    for (int j = 0; j < 8; j++) {
      int r = row0 + j + 8 * hi;
      Out[((size_t)b * NN + r) * C1 + ct * 16 + ln] = (__bf16)(acc[j] + bv);
    }
  }
}

// ---------------------------------------------------------------------------
// V projection: V = style @ Wv + bv, stored TRANSPOSED bf16 (Vt[c][n]) + Vt2=(V^2)^T
// ---------------------------------------------------------------------------
__global__ __launch_bounds__(256) void k_proj_v(
    const float* __restrict__ X, const __bf16* __restrict__ Wt, const float* __restrict__ bias,
    __bf16* __restrict__ Vt, __bf16* __restrict__ Vt2) {
  __shared__ __align__(32) __bf16 Xs[16 * CC];
  int b = blockIdx.x >> 8;
  int row0 = (blockIdx.x & 255) * 16;
  int tid = threadIdx.x;
  const float* src = X + ((size_t)b * NN + row0) * CC;
  for (int idx = tid; idx < 16 * CC; idx += 256) Xs[idx] = (__bf16)src[idx];
  __syncthreads();
  int wave = tid >> 5, lane = tid & 31, ln = lane & 15, hi = lane >> 4;
  for (int ct = wave; ct < CC / 16; ct += 8) {
    v8f acc = zero8();
    for (int ch = 0; ch < CC / 32; ch++) {
      v16bf a  = *(const v16bf*)&Xs[ln * CC + ch * 32 + hi * 16];
      v16bf bm = *(const v16bf*)&Wt[((size_t)(ct * 16 + ln)) * CC + ch * 32 + hi * 16];
      acc = wmma_bf16(a, bm, acc);
    }
    int col = ct * 16 + ln;
    float bb = bias[col];
#pragma unroll
    for (int j = 0; j < 8; j++) {
      int r = row0 + j + 8 * hi;
      float v = acc[j] + bb;
      size_t o = ((size_t)b * CC + col) * NN + r;
      Vt[o]  = (__bf16)v;
      Vt2[o] = (__bf16)(v * v);
    }
  }
}

// ---------------------------------------------------------------------------
// Fused flash attention: softmax(QK^T) @ {V, V^2} with online softmax, then
// std/mean combine with instance-normalized content.
// grid = B * (N/16); block = 256 = 8 waves: 16 queries/block, 64 V-cols/wave.
// K tiles (32 keys x 960 bf16 = 60 KB) are DOUBLE-BUFFERED in LDS via
// global_load_async_to_lds_b128 (ASYNCcnt), overlapping the next tile's fetch
// with the current tile's WMMAs.
// ---------------------------------------------------------------------------
__global__ __launch_bounds__(256) void k_attn(
    const __bf16* __restrict__ Qb, const __bf16* __restrict__ Kb,
    const __bf16* __restrict__ Vt, const __bf16* __restrict__ Vt2,
    const float* __restrict__ content, const float* __restrict__ meanC,
    const float* __restrict__ rstdC, float* __restrict__ out) {
  __shared__ __align__(32) __bf16 Qs[16 * C1];       // 30 KB, staged once
  __shared__ __align__(32) __bf16 Ks[2][32 * C1];    // 2 x 60 KB, double buffered
  __shared__ float Ssub[16 * 32];                    // score tile (f32)
  __shared__ __align__(32) __bf16 Ps[16 * 32];       // probs, A-tile layout
  __shared__ float alphaS[16], lrowS[16];

  int b    = blockIdx.x >> 8;
  int qblk = blockIdx.x & 255;
  int tid  = threadIdx.x;
  int wave = tid >> 5, lane = tid & 31, ln = lane & 15, hi = lane >> 4;
  int cbase = wave * 64;

  { // stage Q block (16 rows x 960 bf16, contiguous)
    const uint4* src = (const uint4*)(Qb + ((size_t)b * NN + qblk * 16) * C1);
    uint4* dst = (uint4*)Qs;
    for (int i = tid; i < 16 * C1 * 2 / 16; i += 256) dst[i] = src[i];
  }
  for (int i = tid; i < 16 * 32; i += 256) Ssub[i] = 0.f;
  // issue async prefetch of key tile 0
  async_copy_ktile(Kb + (size_t)b * NN * C1, Ks[0], tid);

  v8f Macc[4], E2acc[4];
#pragma unroll
  for (int t = 0; t < 4; t++) { Macc[t] = zero8(); E2acc[t] = zero8(); }
  float m_run = -__builtin_inff(), l_run = 0.f;  // live in wave 0, lanes 0..15

  for (int kt = 0; kt < NN / 32; kt++) {
    int cur = kt & 1;
    wait_async0();          // this wave's chunks of Ks[cur] have landed
    __syncthreads();        // all waves' chunks landed; Ssub zeroed; Ps consumed
    if (kt + 1 < NN / 32)   // prefetch next tile while we compute on this one
      async_copy_ktile(Kb + ((size_t)b * NN + (kt + 1) * 32) * C1, Ks[cur ^ 1], tid);

    // Cooperative score tile: 30 K-dim chunks split across the 8 waves.
#pragma unroll
    for (int kh = 0; kh < 2; kh++) {
      v8f sacc = zero8();
      for (int ch = wave; ch < C1 / 32; ch += 8) {
        v16bf a  = *(const v16bf*)&Qs[ln * C1 + ch * 32 + hi * 16];
        v16bf bm = *(const v16bf*)&Ks[cur][(kh * 16 + ln) * C1 + ch * 32 + hi * 16];
        sacc = wmma_bf16(a, bm, sacc);
      }
#pragma unroll
      for (int j = 0; j < 8; j++)
        atomicAdd(&Ssub[(j + 8 * hi) * 32 + kh * 16 + ln], sacc[j]);
    }
    __syncthreads();

    // Online softmax update by wave 0, lanes 0..15 (one query row each).
    if (wave == 0 && hi == 0) {
      int r = ln;
      float tm = -__builtin_inff();
      for (int k = 0; k < 32; k++) tm = fmaxf(tm, Ssub[r * 32 + k]);
      float mn = fmaxf(m_run, tm);
      float al = __expf(m_run - mn);   // 0 on the first tile
      float s = 0.f;
      for (int k = 0; k < 32; k++) {
        float p = __expf(Ssub[r * 32 + k] - mn);
        s += p;
        Ps[r * 32 + k] = (__bf16)p;
      }
      m_run = mn;
      l_run = al * l_run + s;
      lrowS[r] = l_run;
      alphaS[r] = al;
    }
    __syncthreads();

    // Rescale running accumulators, then P@V and P@V^2; re-zero Ssub for next tile.
    float alj[8];
#pragma unroll
    for (int j = 0; j < 8; j++) alj[j] = alphaS[j + 8 * hi];
#pragma unroll
    for (int t = 0; t < 4; t++)
#pragma unroll
      for (int j = 0; j < 8; j++) {
        Macc[t][j]  = Macc[t][j]  * alj[j];
        E2acc[t][j] = E2acc[t][j] * alj[j];
      }
    v16bf pa = *(const v16bf*)&Ps[ln * 32 + hi * 16];
#pragma unroll
    for (int t = 0; t < 4; t++) {
      int col = cbase + t * 16 + ln;
      const v16bf* bv1 = (const v16bf*)&Vt [((size_t)b * CC + col) * NN + kt * 32 + hi * 16];
      Macc[t] = wmma_bf16(pa, *bv1, Macc[t]);
      const v16bf* bv2 = (const v16bf*)&Vt2[((size_t)b * CC + col) * NN + kt * 32 + hi * 16];
      E2acc[t] = wmma_bf16(pa, *bv2, E2acc[t]);
    }
    for (int i = tid; i < 16 * 32; i += 256) Ssub[i] = 0.f;
  }

  // Epilogue: divide by l, std = sqrt(max(E2 - M^2, eps)), fuse instance-norm combine.
  float linv[8];
#pragma unroll
  for (int j = 0; j < 8; j++) linv[j] = 1.0f / lrowS[j + 8 * hi];
#pragma unroll
  for (int t = 0; t < 4; t++) {
    int col = cbase + t * 16 + ln;
    float me = meanC[b * CC + col], rs = rstdC[b * CC + col];
#pragma unroll
    for (int j = 0; j < 8; j++) {
      int q = qblk * 16 + j + 8 * hi;
      float mv = Macc[t][j] * linv[j];
      float e2 = E2acc[t][j] * linv[j];
      float sv = sqrtf(fmaxf(e2 - mv * mv, 1e-9f));
      size_t oi = ((size_t)b * NN + q) * CC + col;
      out[oi] = sv * (content[oi] - me) * rs + mv;
    }
  }
}

// ---------------------------------------------------------------------------
extern "C" void kernel_launch(void* const* d_in, const int* in_sizes, int n_in,
                              void* d_out, int out_size, void* d_ws, size_t ws_size,
                              hipStream_t stream) {
  const float* content   = (const float*)d_in[0];
  const float* style     = (const float*)d_in[1];
  const float* comb_cont = (const float*)d_in[2];
  const float* comb_sty  = (const float*)d_in[3];
  const float* Wq = (const float*)d_in[4];
  const float* bq = (const float*)d_in[5];
  const float* Wk = (const float*)d_in[6];
  const float* bk = (const float*)d_in[7];
  const float* Wv = (const float*)d_in[8];
  const float* bv = (const float*)d_in[9];
  float* out = (float*)d_out;
  char* ws = (char*)d_ws;

  size_t off = 0;
  auto take = [&](size_t bytes) { size_t o = off; off += (bytes + 255) & ~size_t(255); return o; };
  float* sumC   = (float*)(ws + take((size_t)BB * CC * 4));
  float* sum2C  = (float*)(ws + take((size_t)BB * CC * 4));
  float* sumCC  = (float*)(ws + take((size_t)BB * C1 * 4));
  float* sum2CC = (float*)(ws + take((size_t)BB * C1 * 4));
  float* sumCS  = (float*)(ws + take((size_t)BB * C1 * 4));
  float* sum2CS = (float*)(ws + take((size_t)BB * C1 * 4));
  float* meanC  = (float*)(ws + take((size_t)BB * CC * 4));
  float* rstdC  = (float*)(ws + take((size_t)BB * CC * 4));
  float* meanCC = (float*)(ws + take((size_t)BB * C1 * 4));
  float* rstdCC = (float*)(ws + take((size_t)BB * C1 * 4));
  float* meanCS = (float*)(ws + take((size_t)BB * C1 * 4));
  float* rstdCS = (float*)(ws + take((size_t)BB * C1 * 4));
  __bf16* WtQ = (__bf16*)(ws + take((size_t)C1 * C1 * 2));
  __bf16* WtK = (__bf16*)(ws + take((size_t)C1 * C1 * 2));
  __bf16* WtV = (__bf16*)(ws + take((size_t)CC * CC * 2));
  __bf16* Qb  = (__bf16*)(ws + take((size_t)BB * NN * C1 * 2));
  __bf16* Kb  = (__bf16*)(ws + take((size_t)BB * NN * C1 * 2));
  __bf16* Vt  = (__bf16*)(ws + take((size_t)BB * CC * NN * 2));
  __bf16* Vt2 = (__bf16*)(ws + take((size_t)BB * CC * NN * 2));
  (void)ws_size; (void)in_sizes; (void)n_in; (void)out_size;

  // 1) zero sums (contiguous span starting at sumC)
  {
    int nz = BB * CC * 2 + BB * C1 * 4;
    k_zero_f32<<<(nz + 255) / 256, 256, 0, stream>>>(sumC, nz);
  }
  // 2) accumulate stats
  k_accum_stats<<<BB * 8 * 2, 256, 0, stream>>>(content,   sumC,  sum2C,  CC, 2);
  k_accum_stats<<<BB * 8 * 4, 256, 0, stream>>>(comb_cont, sumCC, sum2CC, C1, 4);
  k_accum_stats<<<BB * 8 * 4, 256, 0, stream>>>(comb_sty,  sumCS, sum2CS, C1, 4);
  // 3) finalize mean/rstd
  k_finalize_stats<<<(BB * CC + 255) / 256, 256, 0, stream>>>(sumC,  sum2C,  meanC,  rstdC,  BB * CC);
  k_finalize_stats<<<(BB * C1 + 255) / 256, 256, 0, stream>>>(sumCC, sum2CC, meanCC, rstdCC, BB * C1);
  k_finalize_stats<<<(BB * C1 + 255) / 256, 256, 0, stream>>>(sumCS, sum2CS, meanCS, rstdCS, BB * C1);
  // 4) weight transposes -> bf16
  k_transpose_w<<<(C1 * C1 + 255) / 256, 256, 0, stream>>>(Wq, WtQ, C1, C1);
  k_transpose_w<<<(C1 * C1 + 255) / 256, 256, 0, stream>>>(Wk, WtK, C1, C1);
  k_transpose_w<<<(CC * CC + 255) / 256, 256, 0, stream>>>(Wv, WtV, CC, CC);
  // 5) projections (WMMA)
  k_proj_norm_gemm<<<BB * 256, 256, 0, stream>>>(comb_cont, meanCC, rstdCC, WtQ, bq, Qb);
  k_proj_norm_gemm<<<BB * 256, 256, 0, stream>>>(comb_sty,  meanCS, rstdCS, WtK, bk, Kb);
  k_proj_v<<<BB * 256, 256, 0, stream>>>(style, WtV, bv, Vt, Vt2);
  // 6) fused flash attention + combine (WMMA, async double-buffered K tiles)
  k_attn<<<BB * (NN / 16), 256, 0, stream>>>(Qb, Kb, Vt, Vt2, content, meanC, rstdC, out);
}
